// LSTMEncoder_9174050144552
// MI455X (gfx1250) — matrix operand order repaired and checked by reading, not compiled, not run on hardware
//
#include <hip/hip_runtime.h>

// ---------------------------------------------------------------------------
// AWD-LSTM encoder (eval) for MI455X / gfx1250, wave32, WMMA bf16.
// V=33278, E=400, H=1150, T=128, B=32; M = T*B = 4096 rows.
// Padded dims: Ep=416, Hp=1152 (multiples of 32 for K; gates padded per-gate
// to dhp so 16-wide WMMA N tiles never straddle gate boundaries).
// Round 2: software-pipelined WMMA loops (register double-buffered fragments)
// so next-k loads overlap current WMMAs instead of s_wait_loadcnt 0 per WMMA.
// ---------------------------------------------------------------------------

typedef __attribute__((ext_vector_type(16))) __bf16 v16bf;
typedef __attribute__((ext_vector_type(8)))  __bf16 v8bf;
typedef __attribute__((ext_vector_type(8)))  float  v8f;

#define V_DIM 33278
#define E_DIM 400
#define EP_DIM 416
#define H_DIM 1150
#define HP_DIM 1152
#define T_DIM 128
#define B_DIM 32
#define M_DIM 4096  // T*B

__device__ __forceinline__ unsigned short f32_to_bf16_bits(float f) {
  unsigned int u = __float_as_uint(f);
  unsigned int r = u + 0x7FFFu + ((u >> 16) & 1u);  // round-to-nearest-even
  return (unsigned short)(r >> 16);
}

__device__ __forceinline__ float sigf(float x) {
  return 1.0f / (1.0f + __expf(-x));
}

// A fragment (16x32 bf16): lane l16 = row; group g = lane>>4.
// halfs 0..7  = K[k0 + g*8 .. +7], halfs 8..15 = K[k0+16+g*8 .. +7]
__device__ __forceinline__ v16bf load_frag_a(const unsigned short* __restrict__ row,
                                             int k0, int g) {
  union { v16bf v; v8bf h[2]; } u;
  u.h[0] = *reinterpret_cast<const v8bf*>(row + k0 + g * 8);
  u.h[1] = *reinterpret_cast<const v8bf*>(row + k0 + 16 + g * 8);
  return u.v;
}

// B fragment (32x16 bf16), W stored row-major [N,K] (so B = W^T):
// lane l16 = col n; group g: halfs 0..7 = K[k0 + g*8 .. +7], halfs 8..15 = K[k0+16+g*8 .. +7]
__device__ __forceinline__ v16bf load_frag_b(const unsigned short* __restrict__ row,
                                             int k0, int g) {
  union { v16bf v; v8bf h[2]; } u;
  u.h[0] = *reinterpret_cast<const v8bf*>(row + k0 + g * 8);
  u.h[1] = *reinterpret_cast<const v8bf*>(row + k0 + 16 + g * 8);
  return u.v;
}

__device__ __forceinline__ v8f wmma_bf16(v16bf a, v16bf b, v8f c) {
  return __builtin_amdgcn_wmma_f32_16x16x32_bf16(false, a, false, b,
                                                 (short)0, c, false, false);
}

// ---------------------------------------------------------------------------
// Conversion / packing kernels
// ---------------------------------------------------------------------------

// src [ngate*dh, K] f32 -> dst [ngate*dhp, Kp] bf16, zero-padded rows & cols.
__global__ void pad_convert_kernel(const float* __restrict__ src,
                                   unsigned short* __restrict__ dst,
                                   int dh, int dhp, int ngate, int K, int Kp) {
  size_t i = (size_t)blockIdx.x * blockDim.x + threadIdx.x;
  size_t total = (size_t)ngate * dhp * Kp;
  if (i >= total) return;
  int c = (int)(i % Kp);
  size_t rp = i / Kp;
  int j = (int)(rp % dhp);
  int gate = (int)(rp / dhp);
  float v = 0.0f;
  if (j < dh && c < K) v = src[((size_t)gate * dh + j) * K + c];
  dst[i] = f32_to_bf16_bits(v);
}

// bias = bih + bhh, padded per-gate: dst [4*dhp] f32
__global__ void bias_pad_kernel(const float* __restrict__ bih,
                                const float* __restrict__ bhh,
                                float* __restrict__ dst, int dh, int dhp) {
  int i = blockIdx.x * blockDim.x + threadIdx.x;
  if (i >= 4 * dhp) return;
  int gate = i / dhp, j = i % dhp;
  dst[i] = (j < dh) ? (bih[gate * dh + j] + bhh[gate * dh + j]) : 0.0f;
}

// x0[m, e] = bf16(emb_W[idx[m], e]) with K padding; m = t*B + b
__global__ void embed_gather_kernel(const int* __restrict__ idx,
                                    const float* __restrict__ embW,
                                    unsigned short* __restrict__ x0) {
  size_t i = (size_t)blockIdx.x * blockDim.x + threadIdx.x;
  if (i >= (size_t)M_DIM * EP_DIM) return;
  int e = (int)(i % EP_DIM);
  int m = (int)(i / EP_DIM);
  float v = 0.0f;
  if (e < E_DIM) {
    int row = idx[m];
    v = embW[(size_t)row * E_DIM + e];
  }
  x0[i] = f32_to_bf16_bits(v);
}

// init h (bf16, padded), c (f32, padded), h_st (f32, padded) from h0/c0 [32,dh]
__global__ void init_state_kernel(const float* __restrict__ h0,
                                  const float* __restrict__ c0,
                                  unsigned short* __restrict__ hbf,
                                  float* __restrict__ c_st,
                                  float* __restrict__ h_st, int dh, int dhp) {
  int i = blockIdx.x * blockDim.x + threadIdx.x;
  if (i >= B_DIM * dhp) return;
  int b = i / dhp, j = i % dhp;
  float hv = (j < dh) ? h0[b * dh + j] : 0.0f;
  float cv = (j < dh) ? c0[b * dh + j] : 0.0f;
  hbf[i] = f32_to_bf16_bits(hv);
  c_st[i] = cv;
  h_st[i] = hv;
}

// strip padding: dst[32,dh] <- src[32,dhp]
__global__ void copy_state_kernel(const float* __restrict__ src,
                                  float* __restrict__ dst, int dh, int dhp) {
  int i = blockIdx.x * blockDim.x + threadIdx.x;
  if (i >= B_DIM * dh) return;
  int b = i / dh, j = i % dh;
  dst[i] = src[b * dhp + j];
}

// ---------------------------------------------------------------------------
// GEMM: OUT[M, N] = X[M,Kp](bf16) @ W[N,Kp](bf16)^T + bias[N]   (fp32 out)
// Block: 256 thr = 8 waves; wave tile 16(M) x 64(N); block tile 128 x 64.
// Software-pipelined: next k-step fragments load while current WMMAs issue.
// grid = (ceil(N/64), M/128)
// ---------------------------------------------------------------------------
__global__ __launch_bounds__(256) void gemm_bias_wmma_kernel(
    const unsigned short* __restrict__ X, const unsigned short* __restrict__ W,
    const float* __restrict__ bias, float* __restrict__ OUT,
    int Kp, int Nvalid, int ldout) {
  const int lane = threadIdx.x & 31;
  const int wv = threadIdx.x >> 5;
  const int g = lane >> 4;
  const int l16 = lane & 15;
  const int m_base = blockIdx.y * 128 + wv * 16;
  const int n0 = blockIdx.x * 64;

  const unsigned short* arow = X + (size_t)(m_base + l16) * Kp;
  const unsigned short* brow[4];
#pragma unroll
  for (int j = 0; j < 4; ++j) {
    int n = n0 + j * 16 + l16;
    int nc = (n < Nvalid) ? n : (Nvalid - 1);  // clamp loads; store is masked
    brow[j] = W + (size_t)nc * Kp;
  }

  const v8f vzero = {0.f, 0.f, 0.f, 0.f, 0.f, 0.f, 0.f, 0.f};
  v8f acc[4];
#pragma unroll
  for (int j = 0; j < 4; ++j) acc[j] = vzero;

  // --- software pipeline: prologue loads k=0 ---
  v16bf aC = load_frag_a(arow, 0, g);
  v16bf bC0 = load_frag_b(brow[0], 0, g);
  v16bf bC1 = load_frag_b(brow[1], 0, g);
  v16bf bC2 = load_frag_b(brow[2], 0, g);
  v16bf bC3 = load_frag_b(brow[3], 0, g);

  for (int k0 = 32; k0 < Kp; k0 += 32) {
    // issue next-step loads into fresh registers (stay in flight over WMMAs)
    v16bf aN = load_frag_a(arow, k0, g);
    v16bf bN0 = load_frag_b(brow[0], k0, g);
    v16bf bN1 = load_frag_b(brow[1], k0, g);
    v16bf bN2 = load_frag_b(brow[2], k0, g);
    v16bf bN3 = load_frag_b(brow[3], k0, g);
    acc[0] = wmma_bf16(aC, bC0, acc[0]);
    acc[1] = wmma_bf16(aC, bC1, acc[1]);
    acc[2] = wmma_bf16(aC, bC2, acc[2]);
    acc[3] = wmma_bf16(aC, bC3, acc[3]);
    aC = aN; bC0 = bN0; bC1 = bN1; bC2 = bN2; bC3 = bN3;
  }
  acc[0] = wmma_bf16(aC, bC0, acc[0]);
  acc[1] = wmma_bf16(aC, bC1, acc[1]);
  acc[2] = wmma_bf16(aC, bC2, acc[2]);
  acc[3] = wmma_bf16(aC, bC3, acc[3]);

#pragma unroll
  for (int j = 0; j < 4; ++j) {
    int col = n0 + j * 16 + l16;
    if (col < Nvalid) {
      float bv = bias[col];
#pragma unroll
      for (int r = 0; r < 8; ++r) {
        int row = m_base + r + 8 * g;
        OUT[(size_t)row * ldout + col] = acc[j][r] + bv;
      }
    }
  }
}

// ---------------------------------------------------------------------------
// One LSTM time step, fused: g = xg[t] + h @ Whh^T ; gate math; h/c update.
// Each wave: 2 M-tiles (batch 0..15 / 16..31) x 4 gates x one 16-wide j tile.
// grid = ceil(dhp/128), block 256. h double-buffered (h_in read, h_out write).
// Software-pipelined like the GEMM.
// ---------------------------------------------------------------------------
__global__ __launch_bounds__(256) void lstm_step_wmma_kernel(
    const float* __restrict__ xg, int t,
    const unsigned short* __restrict__ h_in, unsigned short* __restrict__ h_out,
    float* __restrict__ c_st, float* __restrict__ h_st,
    unsigned short* __restrict__ y_out, const unsigned short* __restrict__ Whh,
    int dh, int dhp) {
  const int lane = threadIdx.x & 31;
  const int wv = threadIdx.x >> 5;
  const int g = lane >> 4;
  const int l16 = lane & 15;
  const int j0 = (blockIdx.x * 8 + wv) * 16;
  if (j0 >= dhp) return;  // wave-uniform exit; EXEC stays all-ones for WMMA

  const unsigned short* a0 = h_in + (size_t)l16 * dhp;
  const unsigned short* a1 = h_in + (size_t)(16 + l16) * dhp;
  const unsigned short* brow[4];
#pragma unroll
  for (int gg = 0; gg < 4; ++gg)
    brow[gg] = Whh + (size_t)(gg * dhp + j0 + l16) * dhp;

  const v8f vzero = {0.f, 0.f, 0.f, 0.f, 0.f, 0.f, 0.f, 0.f};
  v8f acc[2][4];
#pragma unroll
  for (int mt = 0; mt < 2; ++mt)
#pragma unroll
    for (int gg = 0; gg < 4; ++gg) acc[mt][gg] = vzero;

  // --- software pipeline ---
  v16bf aC0 = load_frag_a(a0, 0, g);
  v16bf aC1 = load_frag_a(a1, 0, g);
  v16bf bC0 = load_frag_b(brow[0], 0, g);
  v16bf bC1 = load_frag_b(brow[1], 0, g);
  v16bf bC2 = load_frag_b(brow[2], 0, g);
  v16bf bC3 = load_frag_b(brow[3], 0, g);

  for (int k0 = 32; k0 < dhp; k0 += 32) {
    v16bf aN0 = load_frag_a(a0, k0, g);
    v16bf aN1 = load_frag_a(a1, k0, g);
    v16bf bN0 = load_frag_b(brow[0], k0, g);
    v16bf bN1 = load_frag_b(brow[1], k0, g);
    v16bf bN2 = load_frag_b(brow[2], k0, g);
    v16bf bN3 = load_frag_b(brow[3], k0, g);
    acc[0][0] = wmma_bf16(aC0, bC0, acc[0][0]);
    acc[1][0] = wmma_bf16(aC1, bC0, acc[1][0]);
    acc[0][1] = wmma_bf16(aC0, bC1, acc[0][1]);
    acc[1][1] = wmma_bf16(aC1, bC1, acc[1][1]);
    acc[0][2] = wmma_bf16(aC0, bC2, acc[0][2]);
    acc[1][2] = wmma_bf16(aC1, bC2, acc[1][2]);
    acc[0][3] = wmma_bf16(aC0, bC3, acc[0][3]);
    acc[1][3] = wmma_bf16(aC1, bC3, acc[1][3]);
    aC0 = aN0; aC1 = aN1;
    bC0 = bN0; bC1 = bN1; bC2 = bN2; bC3 = bN3;
  }
  acc[0][0] = wmma_bf16(aC0, bC0, acc[0][0]);
  acc[1][0] = wmma_bf16(aC1, bC0, acc[1][0]);
  acc[0][1] = wmma_bf16(aC0, bC1, acc[0][1]);
  acc[1][1] = wmma_bf16(aC1, bC1, acc[1][1]);
  acc[0][2] = wmma_bf16(aC0, bC2, acc[0][2]);
  acc[1][2] = wmma_bf16(aC1, bC2, acc[1][2]);
  acc[0][3] = wmma_bf16(aC0, bC3, acc[0][3]);
  acc[1][3] = wmma_bf16(aC1, bC3, acc[1][3]);

  const int col = j0 + l16;
  const size_t ld4 = (size_t)4 * dhp;
#pragma unroll
  for (int mt = 0; mt < 2; ++mt) {
#pragma unroll
    for (int r = 0; r < 8; ++r) {
      int row = mt * 16 + r + 8 * g;  // batch index 0..31
      const float* xr = xg + ((size_t)t * B_DIM + row) * ld4;
      float iv = acc[mt][0][r] + xr[0 * (size_t)dhp + col];
      float fv = acc[mt][1][r] + xr[1 * (size_t)dhp + col];
      float gv = acc[mt][2][r] + xr[2 * (size_t)dhp + col];
      float ov = acc[mt][3][r] + xr[3 * (size_t)dhp + col];
      size_t sidx = (size_t)row * dhp + col;
      float cn = sigf(fv) * c_st[sidx] + sigf(iv) * tanhf(gv);
      float hn = sigf(ov) * tanhf(cn);
      c_st[sidx] = cn;
      h_st[sidx] = hn;
      unsigned short hb = f32_to_bf16_bits(hn);
      h_out[sidx] = hb;
      y_out[((size_t)t * B_DIM + row) * dhp + col] = hb;
    }
  }
}

// ---------------------------------------------------------------------------
// Host launcher
// ---------------------------------------------------------------------------
extern "C" void kernel_launch(void* const* d_in, const int* in_sizes, int n_in,
                              void* d_out, int out_size, void* d_ws,
                              size_t ws_size, hipStream_t stream) {
  (void)in_sizes; (void)n_in; (void)out_size; (void)ws_size;

  const int* inp = (const int*)d_in[0];
  const float* embW = (const float*)d_in[1];
  const float* finalb = (const float*)d_in[2];
  // per layer l (0..2): d_in[3+6l..]: Wih, Whh, bih, bhh, h0, c0

  float* out = (float*)d_out;

  // ---- workspace carve-up (256B aligned) ----
  char* base = (char*)d_ws;
  size_t off = 0;
  auto alloc = [&](size_t bytes) -> void* {
    void* r = base + off;
    off = (off + bytes + 255) & ~(size_t)255;
    return r;
  };
  unsigned short* emb_bf = (unsigned short*)alloc((size_t)V_DIM * EP_DIM * 2);
  unsigned short* xbuf0  = (unsigned short*)alloc((size_t)M_DIM * HP_DIM * 2);
  unsigned short* xbuf1  = (unsigned short*)alloc((size_t)M_DIM * HP_DIM * 2);
  unsigned short* WA     = (unsigned short*)alloc((size_t)4 * HP_DIM * HP_DIM * 2);
  unsigned short* WB     = (unsigned short*)alloc((size_t)4 * HP_DIM * HP_DIM * 2);
  float*          biasb  = (float*)alloc((size_t)4 * HP_DIM * 4);
  float*          xg     = (float*)alloc((size_t)M_DIM * 4 * HP_DIM * 4);
  unsigned short* hbuf0  = (unsigned short*)alloc((size_t)B_DIM * HP_DIM * 2);
  unsigned short* hbuf1  = (unsigned short*)alloc((size_t)B_DIM * HP_DIM * 2);
  float* c_st[3]; float* h_st[3];
  for (int l = 0; l < 3; ++l) {
    c_st[l] = (float*)alloc((size_t)B_DIM * HP_DIM * 4);
    h_st[l] = (float*)alloc((size_t)B_DIM * HP_DIM * 4);
  }

  // ---- embedding: gather + bf16 convert; tied decoder weight convert ----
  {
    size_t tot = (size_t)V_DIM * EP_DIM;
    pad_convert_kernel<<<(unsigned)((tot + 255) / 256), 256, 0, stream>>>(
        embW, emb_bf, V_DIM, V_DIM, 1, E_DIM, EP_DIM);
    size_t tg = (size_t)M_DIM * EP_DIM;
    embed_gather_kernel<<<(unsigned)((tg + 255) / 256), 256, 0, stream>>>(
        inp, embW, xbuf0);
  }

  struct LCfg { int din, dinp, dh, dhp; };
  const LCfg cfg[3] = {{E_DIM, EP_DIM, H_DIM, HP_DIM},
                       {H_DIM, HP_DIM, H_DIM, HP_DIM},
                       {H_DIM, HP_DIM, E_DIM, EP_DIM}};
  unsigned short* xin[3]  = {xbuf0, xbuf1, xbuf0};
  unsigned short* yout[3] = {xbuf1, xbuf0, xbuf1};

  for (int l = 0; l < 3; ++l) {
    const float* Wih = (const float*)d_in[3 + 6 * l + 0];
    const float* Whh = (const float*)d_in[3 + 6 * l + 1];
    const float* bih = (const float*)d_in[3 + 6 * l + 2];
    const float* bhh = (const float*)d_in[3 + 6 * l + 3];
    const float* h0  = (const float*)d_in[3 + 6 * l + 4];
    const float* c0  = (const float*)d_in[3 + 6 * l + 5];
    const int din = cfg[l].din, dinp = cfg[l].dinp;
    const int dh = cfg[l].dh, dhp = cfg[l].dhp;

    // weight / bias packing
    {
      size_t tA = (size_t)4 * dhp * dinp;
      pad_convert_kernel<<<(unsigned)((tA + 255) / 256), 256, 0, stream>>>(
          Wih, WA, dh, dhp, 4, din, dinp);
      size_t tB = (size_t)4 * dhp * dhp;
      pad_convert_kernel<<<(unsigned)((tB + 255) / 256), 256, 0, stream>>>(
          Whh, WB, dh, dhp, 4, dh, dhp);
      bias_pad_kernel<<<(4 * dhp + 255) / 256, 256, 0, stream>>>(bih, bhh,
                                                                 biasb, dh, dhp);
      init_state_kernel<<<(B_DIM * dhp + 255) / 256, 256, 0, stream>>>(
          h0, c0, hbuf0, c_st[l], h_st[l], dh, dhp);
    }

    // xg[M, 4*dhp] = x @ Wih_p^T + (bih + bhh)   -- batched over all T*B rows
    {
      int N = 4 * dhp;
      dim3 grid((N + 63) / 64, M_DIM / 128);
      gemm_bias_wmma_kernel<<<grid, 256, 0, stream>>>(xin[l], WA, biasb, xg,
                                                      dinp, N, N);
    }

    // sequential recurrence, fused gate math; h double-buffered
    {
      unsigned grid = (unsigned)((dhp + 127) / 128);
      for (int t = 0; t < T_DIM; ++t) {
        unsigned short* hi = (t & 1) ? hbuf1 : hbuf0;
        unsigned short* ho = (t & 1) ? hbuf0 : hbuf1;
        lstm_step_wmma_kernel<<<grid, 256, 0, stream>>>(
            xg, t, hi, ho, c_st[l], h_st[l], yout[l], WB, dh, dhp);
      }
    }
  }

  // ---- tied decoder: logits[M, V] = y2 @ emb^T + final_b ----
  {
    dim3 grid((V_DIM + 63) / 64, M_DIM / 128);
    gemm_bias_wmma_kernel<<<grid, 256, 0, stream>>>(xbuf1, emb_bf, finalb, out,
                                                    EP_DIM, V_DIM, V_DIM);
  }

  // ---- final hidden states into d_out tail ----
  {
    size_t o = (size_t)M_DIM * V_DIM;
    const int dhs[3] = {H_DIM, H_DIM, E_DIM};
    const int dhps[3] = {HP_DIM, HP_DIM, EP_DIM};
    for (int l = 0; l < 3; ++l) {
      int n = B_DIM * dhs[l];
      copy_state_kernel<<<(n + 255) / 256, 256, 0, stream>>>(h_st[l], out + o,
                                                             dhs[l], dhps[l]);
      o += n;
      copy_state_kernel<<<(n + 255) / 256, 256, 0, stream>>>(c_st[l], out + o,
                                                             dhs[l], dhps[l]);
      o += n;
    }
  }
}
